// SchNet_59030030516409
// MI455X (gfx1250) — matrix operand order repaired and checked by reading, not compiled, not run on hardware
//
#include <hip/hip_runtime.h>

// ---------------- problem constants ----------------
static constexpr int kN  = 20000;
static constexpr int kE  = 320000;
static constexpr int kG  = 64;
static constexpr int kH  = 128;
static constexpr int kF  = 128;
static constexpr int kNG = 50;     // gaussians
static constexpr int kNGP= 64;     // padded
static constexpr int kNI = 6;
#define CUTF   5.0f
#define PI_F   3.14159265358979323846f
#define DELTA_F (5.0f/49.0f)
#define GAMMA_F (0.5f/(DELTA_F*DELTA_F))
#define LOG2_F 0.6931471805599453f
static constexpr int STR = 132;    // LDS row stride in halves (pad vs 128)

typedef __attribute__((ext_vector_type(16))) _Float16 v16h;
typedef __attribute__((ext_vector_type(8)))  float    v8f;
typedef __attribute__((ext_vector_type(2)))  _Float16 h2;

// ---------------- device helpers ----------------
__device__ __forceinline__ float sspf(float x) {           // softplus(x)-log2
  float sp = (x > 20.f) ? x : log1pf(expf(x));
  return sp - LOG2_F;
}
__device__ __forceinline__ float sigf(float x) { return 1.f/(1.f+expf(-x)); }

__device__ __forceinline__ void atomAddF(float* p, float v) { unsafeAtomicAdd(p, v); }

// A-operand (16x32 f16) from f32 row-major source. m = lane%16, half = lane>>4.
// VGPR pair p holds K = kbase + (p<4 ? 2p : 8+2p) + 8*half  (+1 for hi element)
template<int ACT>
__device__ __forceinline__ v16h ldA_f32(const float* A, int ld, int m, int kbase, int half) {
  v16h r;
#pragma unroll
  for (int p = 0; p < 8; ++p) {
    int k = kbase + ((p < 4) ? 2*p : 8 + 2*p) + 8*half;
    float2 f = *(const float2*)(A + (size_t)m*ld + k);
    float x = f.x, y = f.y;
    if (ACT) { x = sspf(x); y = sspf(y); }
    r[2*p] = (_Float16)x; r[2*p+1] = (_Float16)y;
  }
  return r;
}
__device__ __forceinline__ v16h ldA_f16(const _Float16* A, int ld, int m, int kbase, int half) {
  v16h r;
#pragma unroll
  for (int p = 0; p < 8; ++p) {
    int k = kbase + ((p < 4) ? 2*p : 8 + 2*p) + 8*half;
    h2 x = *(const h2*)(A + (size_t)m*ld + k);
    r[2*p] = x[0]; r[2*p+1] = x[1];
  }
  return r;
}
// B-operand (32x16 f16) from transposed weight Wt[N][Kpad] (row-major).
// lane n = lane%16; pair p holds K = kbase + (p<4 ? 2p : 8+2p) + 8*half (+1)
__device__ __forceinline__ v16h ldB_f16(const _Float16* Wt, int ldk, int n, int kbase, int half) {
  v16h r;
#pragma unroll
  for (int p = 0; p < 8; ++p) {
    int k = kbase + ((p < 4) ? 2*p : 8 + 2*p) + 8*half;
    h2 x = *(const h2*)(Wt + (size_t)n*ldk + k);
    r[2*p] = x[0]; r[2*p+1] = x[1];
  }
  return r;
}
__device__ __forceinline__ v8f wmma16(v16h a, v16h b, v8f c) {
  return __builtin_amdgcn_wmma_f32_16x16x32_f16(false, a, false, b, (short)0, c, false, false);
}

// ---------------- small kernels ----------------
__global__ void convW(const float* __restrict__ W, _Float16* __restrict__ Wrm,
                      _Float16* __restrict__ Wt, int K, int Kpad, int Nc) {
  int idx = blockIdx.x*blockDim.x + threadIdx.x;
  if (idx >= Kpad*Nc) return;
  int k = idx / Nc, n = idx % Nc;
  float v = (k < K) ? W[(size_t)k*Nc + n] : 0.f;
  Wrm[(size_t)k*Nc + n] = (_Float16)v;
  Wt[(size_t)n*Kpad + k] = (_Float16)v;
}

__global__ void edge_geom(const float* __restrict__ pos, const int* __restrict__ row,
                          const int* __restrict__ col, float* __restrict__ dE,
                          float* __restrict__ uE, _Float16* __restrict__ rbf16, int Ec) {
  int e = blockIdx.x*blockDim.x + threadIdx.x;
  if (e >= Ec) return;
  int r = row[e], c = col[e];
  float dx = pos[3*r+0]-pos[3*c+0], dy = pos[3*r+1]-pos[3*c+1], dz = pos[3*r+2]-pos[3*c+2];
  float d = sqrtf(dx*dx + dy*dy + dz*dz);
  float inv = (d > 1e-12f) ? 1.f/d : 0.f;
  dE[e] = d; uE[e] = dx*inv; uE[Ec+e] = dy*inv; uE[2*Ec+e] = dz*inv;
#pragma unroll
  for (int g = 0; g < kNGP; g += 2) {              // packed f16 pair stores
    float v0 = 0.f, v1 = 0.f;
    if (g   < kNG) { float t = d - g*DELTA_F;     v0 = expf(-GAMMA_F*t*t); }
    if (g+1 < kNG) { float t = d - (g+1)*DELTA_F; v1 = expf(-GAMMA_F*t*t); }
    h2 pk; pk[0] = (_Float16)v0; pk[1] = (_Float16)v1;
    *(h2*)(rbf16 + (size_t)e*kNGP + g) = pk;
  }
}

__global__ void embed_k(const int* __restrict__ z, const float* __restrict__ emb,
                        float* __restrict__ h0, int n) {
  int idx = blockIdx.x*blockDim.x + threadIdx.x;
  if (idx >= n) return;
  int a = idx >> 7, c = idx & 127;
  h0[idx] = emb[(size_t)z[a]*kH + c];
}

__global__ void head_fwd(const float* __restrict__ u1, const float* __restrict__ ow2,
                         const float* __restrict__ ob2, const int* __restrict__ batch,
                         float* __restrict__ outE, int n) {
  int i = blockIdx.x*blockDim.x + threadIdx.x;
  if (i >= n) return;
  float acc = ob2[0];
#pragma unroll 8
  for (int j = 0; j < 64; ++j) acc += sspf(u1[(size_t)i*64 + j]) * ow2[j];
  atomAddF(outE + batch[i], acc);
}

__global__ void ew_du1(float* __restrict__ u1, const float* __restrict__ ow2, int n) {
  int idx = blockIdx.x*blockDim.x + threadIdx.x;
  if (idx >= n) return;
  u1[idx] = sigf(u1[idx]) * ow2[idx & 63];
}

__global__ void ew_mulsig(float* __restrict__ t, const float* __restrict__ hc, int n) {
  int idx = blockIdx.x*blockDim.x + threadIdx.x;
  if (idx >= n) return;
  t[idx] *= sigf(hc[idx]);
}

__global__ void write_forces(const float* __restrict__ gpos, float* __restrict__ out, int n) {
  int idx = blockIdx.x*blockDim.x + threadIdx.x;
  if (idx >= n) return;
  out[kG + idx] = -gpos[idx];
}

// ---------------- node GEMM: C[M,NC] = act(A)[M,K] @ W + bias + add ----------------
template<int K, int NC, int ACT>
__global__ __launch_bounds__(128) void gemm_rm(const float* __restrict__ A,
                                               const _Float16* __restrict__ Wt,
                                               const float* __restrict__ bias,
                                               const float* __restrict__ add,
                                               float* __restrict__ C, int M) {
  constexpr int NT = NC/16, KC = K/32;
  int wave = threadIdx.x >> 5, lane = threadIdx.x & 31;
  int half = lane >> 4, q = lane & 15;
  int tile = blockIdx.x*4 + wave;
  if (tile*16 >= M) return;
  int m0 = tile*16;
  v8f acc[NT];
#pragma unroll
  for (int t = 0; t < NT; ++t)
#pragma unroll
    for (int r = 0; r < 8; ++r) acc[t][r] = 0.f;
  const float* Ab = A + (size_t)m0*K;
#pragma unroll
  for (int kc = 0; kc < KC; ++kc) {
    v16h a = ldA_f32<ACT>(Ab, K, q, kc*32, half);
#pragma unroll
    for (int t = 0; t < NT; ++t) {
      v16h b = ldB_f16(Wt, K, t*16+q, kc*32, half);
      acc[t] = wmma16(a, b, acc[t]);
    }
  }
#pragma unroll
  for (int t = 0; t < NT; ++t) {
    int nn = t*16 + q;
    float bn = bias ? bias[nn] : 0.f;
#pragma unroll
    for (int r = 0; r < 8; ++r) {
      int mm = m0 + r + 8*half;
      float v = acc[t][r] + bn;
      if (add) v += add[(size_t)mm*NC + nn];
      C[(size_t)mm*NC + nn] = v;
    }
  }
}

// ---------------- fused edge filter + message + scatter (forward) ----------------
__global__ __launch_bounds__(128) void edge_fwd(const _Float16* __restrict__ rbf16,
    const _Float16* __restrict__ w1T, const float* __restrict__ b1,
    const _Float16* __restrict__ w2T, const float* __restrict__ b2,
    const float* __restrict__ dE, const float* __restrict__ hf,
    const int* __restrict__ row, const int* __restrict__ col, float* __restrict__ agg) {
  __shared__ _Float16 sm[4*16*STR];
  int wave = threadIdx.x >> 5, lane = threadIdx.x & 31;
  int half = lane >> 4, q = lane & 15;
  int e0 = (blockIdx.x*4 + wave)*16;
  _Float16* S = sm + wave*16*STR;

  // Load per-edge indices up front and prefetch the gather targets so the
  // random hf-row accesses resolve in L2 while the two GEMMs run.
  float cf[8]; int ci[8], ri[8];
#pragma unroll
  for (int r = 0; r < 8; ++r) {
    int e = e0 + r + 8*half;
    float d = dE[e];
    cf[r] = 0.5f*(cosf(d*(PI_F/CUTF)) + 1.f);
    ci[r] = col[e]; ri[r] = row[e];
    __builtin_prefetch(hf + (size_t)ci[r]*kF + q*8, 0, 1);   // 16 lanes span the 512B row
  }

  v8f c[8];
#pragma unroll
  for (int t = 0; t < 8; ++t)
#pragma unroll
    for (int r = 0; r < 8; ++r) c[t][r] = 0.f;
  const _Float16* Ab = rbf16 + (size_t)e0*kNGP;
#pragma unroll
  for (int kc = 0; kc < 2; ++kc) {                 // K = 64 (padded RBF)
    v16h a = ldA_f16(Ab, kNGP, q, kc*32, half);
#pragma unroll
    for (int t = 0; t < 8; ++t) { v16h b = ldB_f16(w1T, kNGP, t*16+q, kc*32, half); c[t] = wmma16(a,b,c[t]); }
  }
#pragma unroll
  for (int t = 0; t < 8; ++t) {
    float bn = b1[t*16+q];
#pragma unroll
    for (int r = 0; r < 8; ++r)
      S[(r+8*half)*STR + t*16 + q] = (_Float16)sspf(c[t][r] + bn);
  }
#pragma unroll
  for (int t = 0; t < 8; ++t)
#pragma unroll
    for (int r = 0; r < 8; ++r) c[t][r] = 0.f;
#pragma unroll
  for (int kc = 0; kc < 4; ++kc) {                 // K = 128
    v16h a = ldA_f16(S, STR, q, kc*32, half);
#pragma unroll
    for (int t = 0; t < 8; ++t) { v16h b = ldB_f16(w2T, kF, t*16+q, kc*32, half); c[t] = wmma16(a,b,c[t]); }
  }
#pragma unroll
  for (int t = 0; t < 8; ++t) {
    int nn = t*16 + q; float bn = b2[nn];
#pragma unroll
    for (int r = 0; r < 8; ++r) {
      float f = c[t][r] + bn;
      float m = f * cf[r] * hf[(size_t)ci[r]*kF + nn];
      atomAddF(agg + (size_t)ri[r]*kF + nn, m);
    }
  }
}

// ---------------- fused edge backward: recompute filter net, backprop to dhf, d, pos ----------------
__global__ __launch_bounds__(128) void edge_bwd(const _Float16* __restrict__ rbf16,
    const _Float16* __restrict__ w1T, const float* __restrict__ b1,
    const _Float16* __restrict__ w2T, const float* __restrict__ b2,
    const _Float16* __restrict__ w1rm, const _Float16* __restrict__ w2rm,
    const float* __restrict__ dE, const float* __restrict__ uE,
    const float* __restrict__ hf, const float* __restrict__ dagg,
    const int* __restrict__ row, const int* __restrict__ col,
    float* __restrict__ dhf, float* __restrict__ gpos, int Ec) {
  __shared__ _Float16 sm[4*2*16*STR];
  int wave = threadIdx.x >> 5, lane = threadIdx.x & 31;
  int half = lane >> 4, q = lane & 15;
  int e0 = (blockIdx.x*4 + wave)*16;
  _Float16* S0 = sm + wave*(2*16*STR);
  _Float16* S1 = S0 + 16*STR;

  float cf[8], dv[8]; int ci[8], ri[8];
#pragma unroll
  for (int r = 0; r < 8; ++r) {
    int e = e0 + r + 8*half;
    float d = dE[e]; dv[r] = d;
    cf[r] = 0.5f*(cosf(d*(PI_F/CUTF)) + 1.f);
    ci[r] = col[e]; ri[r] = row[e];
    __builtin_prefetch(dagg + (size_t)ri[r]*kF + q*8, 0, 1); // prefetch gather rows
    __builtin_prefetch(hf   + (size_t)ci[r]*kF + q*8, 0, 1);
  }

  v8f c[8];
#pragma unroll
  for (int t = 0; t < 8; ++t)
#pragma unroll
    for (int r = 0; r < 8; ++r) c[t][r] = 0.f;
  const _Float16* Ab = rbf16 + (size_t)e0*kNGP;
#pragma unroll
  for (int kc = 0; kc < 2; ++kc) {                 // GEMM1: t1 = rbf @ mw1
    v16h a = ldA_f16(Ab, kNGP, q, kc*32, half);
#pragma unroll
    for (int t = 0; t < 8; ++t) { v16h b = ldB_f16(w1T, kNGP, t*16+q, kc*32, half); c[t] = wmma16(a,b,c[t]); }
  }
#pragma unroll
  for (int t = 0; t < 8; ++t) {
    float bn = b1[t*16+q];
#pragma unroll
    for (int r = 0; r < 8; ++r) {
      float t1 = c[t][r] + bn;
      int p = (r+8*half)*STR + t*16 + q;
      S0[p] = (_Float16)sspf(t1);                  // s1 for GEMM2
      S1[p] = (_Float16)sigf(t1);                  // ssp' for dt1
    }
  }
#pragma unroll
  for (int t = 0; t < 8; ++t)
#pragma unroll
    for (int r = 0; r < 8; ++r) c[t][r] = 0.f;
#pragma unroll
  for (int kc = 0; kc < 4; ++kc) {                 // GEMM2: filt = s1 @ mw2
    v16h a = ldA_f16(S0, STR, q, kc*32, half);
#pragma unroll
    for (int t = 0; t < 8; ++t) { v16h b = ldB_f16(w2T, kF, t*16+q, kc*32, half); c[t] = wmma16(a,b,c[t]); }
  }
  float pc[8];
#pragma unroll
  for (int r = 0; r < 8; ++r) pc[r] = 0.f;
#pragma unroll
  for (int t = 0; t < 8; ++t) {
    int nn = t*16 + q; float bn = b2[nn];
#pragma unroll
    for (int r = 0; r < 8; ++r) {
      float filt = c[t][r] + bn;
      float dm  = dagg[(size_t)ri[r]*kF + nn];     // dmsg = dagg[row]
      float hv  = hf[(size_t)ci[r]*kF + nn];
      pc[r] += dm * filt * hv;                     // d(cfac) partial
      atomAddF(dhf + (size_t)ci[r]*kF + nn, dm * cf[r] * filt);
      S0[(r+8*half)*STR + nn] = (_Float16)(dm * cf[r] * hv);   // dfilt (A for next GEMM)
    }
  }
#pragma unroll
  for (int t = 0; t < 8; ++t)
#pragma unroll
    for (int r = 0; r < 8; ++r) c[t][r] = 0.f;
#pragma unroll
  for (int kc = 0; kc < 4; ++kc) {                 // ds1 = dfilt @ mw2^T
    v16h a = ldA_f16(S0, STR, q, kc*32, half);
#pragma unroll
    for (int t = 0; t < 8; ++t) { v16h b = ldB_f16(w2rm, kF, t*16+q, kc*32, half); c[t] = wmma16(a,b,c[t]); }
  }
#pragma unroll
  for (int t = 0; t < 8; ++t)
#pragma unroll
    for (int r = 0; r < 8; ++r) {                  // dt1 = ds1 * sigmoid(t1)
      int p = (r+8*half)*STR + t*16 + q;
      S0[p] = (_Float16)(c[t][r] * (float)S1[p]);
    }
  v8f c4[4];
#pragma unroll
  for (int t = 0; t < 4; ++t)
#pragma unroll
    for (int r = 0; r < 8; ++r) c4[t][r] = 0.f;
#pragma unroll
  for (int kc = 0; kc < 4; ++kc) {                 // drbf = dt1 @ mw1^T  [16 x 64]
    v16h a = ldA_f16(S0, STR, q, kc*32, half);
#pragma unroll
    for (int t = 0; t < 4; ++t) { v16h b = ldB_f16(w1rm, kF, t*16+q, kc*32, half); c4[t] = wmma16(a,b,c4[t]); }
  }
  float pd[8];
#pragma unroll
  for (int r = 0; r < 8; ++r) pd[r] = 0.f;
#pragma unroll
  for (int t = 0; t < 4; ++t) {
    int g = t*16 + q; float mu = g*DELTA_F;
#pragma unroll
    for (int r = 0; r < 8; ++r) {
      int e = e0 + r + 8*half;
      float rb = (float)rbf16[(size_t)e*kNGP + g];
      pd[r] += c4[t][r] * (-2.f*GAMMA_F) * (dv[r] - mu) * rb;
    }
  }
#pragma unroll
  for (int mask = 1; mask < 16; mask <<= 1) {      // reduce across the 16 lanes of each half
#pragma unroll
    for (int r = 0; r < 8; ++r) {
      pc[r] += __shfl_xor(pc[r], mask, 32);
      pd[r] += __shfl_xor(pd[r], mask, 32);
    }
  }
#pragma unroll
  for (int r = 0; r < 8; ++r) {
    if (q == r) {
      int e = e0 + r + 8*half;
      float d = dv[r];
      float dd = pd[r] + pc[r] * (-0.5f*(PI_F/CUTF)) * sinf(d*(PI_F/CUTF));
      float ux = uE[e], uy = uE[Ec+e], uz = uE[2*Ec+e];
      int R = ri[r], Cc = ci[r];
      atomAddF(gpos + 3*R + 0,  dd*ux);
      atomAddF(gpos + 3*R + 1,  dd*uy);
      atomAddF(gpos + 3*R + 2,  dd*uz);
      atomAddF(gpos + 3*Cc + 0, -dd*ux);
      atomAddF(gpos + 3*Cc + 1, -dd*uy);
      atomAddF(gpos + 3*Cc + 2, -dd*uz);
    }
  }
}

// ---------------- host orchestration ----------------
extern "C" void kernel_launch(void* const* d_in, const int* in_sizes, int n_in,
                              void* d_out, int out_size, void* d_ws, size_t ws_size,
                              hipStream_t stream) {
  (void)in_sizes; (void)n_in; (void)out_size; (void)ws_size;
  const int*   z    = (const int*)  d_in[0];
  const float* pos  = (const float*)d_in[1];
  const int*   ei   = (const int*)  d_in[2];
  const int*   batch= (const int*)  d_in[3];
  const float* emb  = (const float*)d_in[4];
  const float* mw1  = (const float*)d_in[5];
  const float* mb1  = (const float*)d_in[6];
  const float* mw2  = (const float*)d_in[7];
  const float* mb2  = (const float*)d_in[8];
  const float* cw1  = (const float*)d_in[9];
  const float* cw2  = (const float*)d_in[10];
  const float* cb2  = (const float*)d_in[11];
  const float* lw   = (const float*)d_in[12];
  const float* lb   = (const float*)d_in[13];
  const float* ow1  = (const float*)d_in[14];
  const float* ob1  = (const float*)d_in[15];
  const float* ow2  = (const float*)d_in[16];
  const float* ob2  = (const float*)d_in[17];
  float* out = (float*)d_out;
  const int* row = ei;
  const int* col = ei + kE;

  char* ws = (char*)d_ws;
  size_t off = 0;
  auto take = [&](size_t bytes) -> char* {
    char* p = ws + off; off = (off + bytes + 255) & ~(size_t)255; return p;
  };
  _Float16 *mw1rm[kNI], *mw1T[kNI], *mw2rm[kNI], *mw2T[kNI];
  _Float16 *cw1rm[kNI], *cw1T[kNI], *cw2rm[kNI], *cw2T[kNI], *lwrm[kNI], *lwT[kNI];
  for (int i = 0; i < kNI; ++i) {
    mw1rm[i] = (_Float16*)take(kNGP*kF*2);  mw1T[i] = (_Float16*)take(kF*kNGP*2);
    mw2rm[i] = (_Float16*)take(kF*kF*2);    mw2T[i] = (_Float16*)take(kF*kF*2);
    cw1rm[i] = (_Float16*)take(kH*kF*2);    cw1T[i] = (_Float16*)take(kF*kH*2);
    cw2rm[i] = (_Float16*)take(kF*kH*2);    cw2T[i] = (_Float16*)take(kH*kF*2);
    lwrm[i]  = (_Float16*)take(kH*kH*2);    lwT[i]  = (_Float16*)take(kH*kH*2);
  }
  _Float16* ow1rm = (_Float16*)take(kH*64*2);
  _Float16* ow1T  = (_Float16*)take(64*kH*2);
  _Float16* rbf16 = (_Float16*)take((size_t)kE*kNGP*2);
  float* dEbuf = (float*)take((size_t)kE*4);
  float* uE    = (float*)take((size_t)3*kE*4);
  float* hh[kNI+1];
  for (int i = 0; i <= kNI; ++i) hh[i] = (float*)take((size_t)kN*kH*4);
  float* hch[kNI];
  for (int i = 0; i < kNI; ++i) hch[i] = (float*)take((size_t)kN*kF*4);
  float* hfb  = (float*)take((size_t)kN*kF*4);
  float* agg  = (float*)take((size_t)kN*kF*4);
  float* u1   = (float*)take((size_t)kN*64*4);
  float* dhA  = (float*)take((size_t)kN*kH*4);
  float* dhB  = (float*)take((size_t)kN*kH*4);
  float* t1b  = (float*)take((size_t)kN*kF*4);
  float* t2b  = (float*)take((size_t)kN*kF*4);
  float* dhfb = (float*)take((size_t)kN*kF*4);
  float* gpos = (float*)take((size_t)kN*3*4);

  const int cB = 256;
  auto cgrid = [](int n, int b){ return (n + b - 1)/b; };

  // weight conversion (f16 row-major + transposed copies)
  for (int i = 0; i < kNI; ++i) {
    convW<<<cgrid(kNGP*kF,cB),cB,0,stream>>>(mw1 + (size_t)i*kNG*kF, mw1rm[i], mw1T[i], kNG, kNGP, kF);
    convW<<<cgrid(kF*kF,cB),cB,0,stream>>>(mw2 + (size_t)i*kF*kF, mw2rm[i], mw2T[i], kF, kF, kF);
    convW<<<cgrid(kH*kF,cB),cB,0,stream>>>(cw1 + (size_t)i*kH*kF, cw1rm[i], cw1T[i], kH, kH, kF);
    convW<<<cgrid(kF*kH,cB),cB,0,stream>>>(cw2 + (size_t)i*kF*kH, cw2rm[i], cw2T[i], kF, kF, kH);
    convW<<<cgrid(kH*kH,cB),cB,0,stream>>>(lw  + (size_t)i*kH*kH, lwrm[i],  lwT[i],  kH, kH, kH);
  }
  convW<<<cgrid(kH*64,cB),cB,0,stream>>>(ow1, ow1rm, ow1T, kH, kH, 64);

  edge_geom<<<cgrid(kE,cB),cB,0,stream>>>(pos, row, col, dEbuf, uE, rbf16, kE);
  embed_k<<<cgrid(kN*kH,cB),cB,0,stream>>>(z, emb, hh[0], kN*kH);

  const int gN = (kN/16 + 3)/4;     // node GEMM grid (1250 tiles / 4 waves)
  const int gEdge = kE/16/4;        // 5000 blocks, 4 waves each

  // ---------------- forward ----------------
  for (int i = 0; i < kNI; ++i) {
    gemm_rm<128,128,0><<<gN,128,0,stream>>>(hh[i], cw1T[i], nullptr, nullptr, hfb, kN);
    hipMemsetAsync(agg, 0, (size_t)kN*kF*4, stream);
    edge_fwd<<<gEdge,128,0,stream>>>(rbf16, mw1T[i], mb1 + (size_t)i*kF, mw2T[i],
                                     mb2 + (size_t)i*kF, dEbuf, hfb, row, col, agg);
    gemm_rm<128,128,0><<<gN,128,0,stream>>>(agg, cw2T[i], cb2 + (size_t)i*kH, nullptr, hch[i], kN);
    gemm_rm<128,128,1><<<gN,128,0,stream>>>(hch[i], lwT[i], lb + (size_t)i*kH, hh[i], hh[i+1], kN);
  }
  gemm_rm<128,64,0><<<gN,128,0,stream>>>(hh[kNI], ow1T, ob1, nullptr, u1, kN);
  hipMemsetAsync(out, 0, kG*sizeof(float), stream);
  head_fwd<<<cgrid(kN,cB),cB,0,stream>>>(u1, ow2, ob2, batch, out, kN);

  // ---------------- backward (forces) ----------------
  ew_du1<<<cgrid(kN*64,cB),cB,0,stream>>>(u1, ow2, kN*64);
  gemm_rm<64,128,0><<<gN,128,0,stream>>>(u1, ow1rm, nullptr, nullptr, dhA, kN);
  hipMemsetAsync(gpos, 0, (size_t)kN*3*4, stream);

  float* dIn = dhA; float* dOut = dhB;
  for (int i = kNI - 1; i >= 0; --i) {
    gemm_rm<128,128,0><<<gN,128,0,stream>>>(dIn, lwrm[i], nullptr, nullptr, t1b, kN);
    ew_mulsig<<<cgrid(kN*kF,cB),cB,0,stream>>>(t1b, hch[i], kN*kF);           // dhc
    gemm_rm<128,128,0><<<gN,128,0,stream>>>(t1b, cw2rm[i], nullptr, nullptr, t2b, kN); // dagg
    gemm_rm<128,128,0><<<gN,128,0,stream>>>(hh[i], cw1T[i], nullptr, nullptr, hfb, kN); // hf recompute
    hipMemsetAsync(dhfb, 0, (size_t)kN*kF*4, stream);
    edge_bwd<<<gEdge,128,0,stream>>>(rbf16, mw1T[i], mb1 + (size_t)i*kF, mw2T[i],
                                     mb2 + (size_t)i*kF, mw1rm[i], mw2rm[i],
                                     dEbuf, uE, hfb, t2b, row, col, dhfb, gpos, kE);
    gemm_rm<128,128,0><<<gN,128,0,stream>>>(dhfb, cw1rm[i], nullptr, dIn, dOut, kN);
    float* tmp = dIn; dIn = dOut; dOut = tmp;
  }
  write_forces<<<cgrid(kN*3,cB),cB,0,stream>>>(gpos, out, kN*3);
}
